// _NonLocalNDEmbeddedGaussian_41910290874570
// MI455X (gfx1250) — matrix operand order repaired and checked by reading, not compiled, not run on hardware
//
#include <hip/hip_runtime.h>
#include <math.h>

// ---------------- problem constants ----------------
constexpr int kN  = 8;
constexpr int kC  = 512;
constexpr int kH  = 56;
constexpr int kW  = 56;
constexpr int kCI = 256;
constexpr int kT  = kH * kW;      // 3136
constexpr int kHP = kH / 2;       // 28
constexpr int kWP = kW / 2;       // 28
constexpr int kP  = kHP * kWP;    // 784
// padded dims: every GEMM gets M%64==0, N%128==0, K%32==0 -> zero guards
constexpr int kTp = 3200;         // 25*128
constexpr int kPp = 896;          // 7*128 = 28*32

// ---------------- native vector + global (AS1) pointer typedefs ----------------
typedef unsigned int v2u __attribute__((ext_vector_type(2)));
typedef __attribute__((address_space(1))) const unsigned short gc_ushort;
typedef __attribute__((address_space(1))) const v2u            gc_v2u;
typedef __attribute__((address_space(1))) const float          gc_float;
typedef __attribute__((address_space(1))) float                g_float;
typedef __attribute__((address_space(1))) unsigned short       g_ushort;

// ---------------- bf16 helpers (raw ushort storage) ----------------
__device__ __forceinline__ unsigned short f32_to_bf16(float f) {
  unsigned int u = __float_as_uint(f);
  unsigned int r = u + 0x7FFFu + ((u >> 16) & 1u);   // round-to-nearest-even
  return (unsigned short)(r >> 16);
}

// ---------------- WMMA fragment types ----------------
typedef __bf16 v16bf __attribute__((ext_vector_type(16)));
typedef float  v8f   __attribute__((ext_vector_type(8)));

union FragAB { unsigned int u[8]; v16bf v; };
union FragD  { v8f v; float f[8]; };

// ---------------- generic bf16 WMMA GEMM (guard-free, LDS double-buffered) ---
// C[M,N] = scale * (A[M,K] * B[K,N]) + bias[m]
// Requires M%64==0, N%128==0, K%32==0, lda/ldb multiples of 4.
// Output modes: 0: f32 out[m*ldc+n]   2: bf16 out[n*ldc+m] (transposed store)
#define BM 64
#define BN 128
#define BK 32

__global__ __launch_bounds__(256) void gemm_bf16_wmma(
    const unsigned short* __restrict__ A, const unsigned short* __restrict__ B,
    const float* __restrict__ bias, void* __restrict__ outv,
    int M, int Nn, int K, int lda, int ldb, int ldc,
    float scale, int mode,
    long long strideA, long long strideB, long long strideOutBytes)
{
  __shared__ __align__(16) unsigned short lds_a[2][BM * BK];  // [m][k]
  __shared__ __align__(16) unsigned short lds_b[2][BN * BK];  // [n][k] (transposed)

  const int tid  = threadIdx.x;
  const int lane = tid & 31;
  const int wave = tid >> 5;
  const int wm   = (wave >> 2) * 32;       // 0 or 32
  const int wn   = (wave & 3)  * 32;       // 0..96
  const int bm0  = blockIdx.y * BM;
  const int bn0  = blockIdx.x * BN;

  // uniform global bases (scalar regs) + 32-bit per-lane offsets
  gc_ushort* Ag = (gc_ushort*)(A + (long long)blockIdx.z * strideA);
  gc_ushort* Bg = (gc_ushort*)(B + (long long)blockIdx.z * strideB);
  char* outb = (char*)outv + (long long)blockIdx.z * strideOutBytes;

  // coop-load geometry: quads of 4 bf16 per thread
  // A tile: 64x32 = 512 quads -> 2 per thread
  const int aq0 = tid, aq1 = tid + 256;
  unsigned int aoff0 = (unsigned)(bm0 + (aq0 >> 3)) * lda + (aq0 & 7) * 4;
  unsigned int aoff1 = (unsigned)(bm0 + (aq1 >> 3)) * lda + (aq1 & 7) * 4;
  // B tile: 32x128 = 1024 quads -> 4 per thread
  int bkk[4], bnn[4];
  unsigned int boff[4];
#pragma unroll
  for (int j = 0; j < 4; ++j) {
    int q = tid + j * 256;
    bkk[j] = q >> 5;
    bnn[j] = (q & 31) * 4;
    boff[j] = (unsigned)bkk[j] * ldb + bn0 + bnn[j];
  }

  v8f zero;
#pragma unroll
  for (int r = 0; r < 8; ++r) zero[r] = 0.0f;
  v8f acc[2][2];
  acc[0][0] = zero; acc[0][1] = zero; acc[1][0] = zero; acc[1][1] = zero;

  const int mrow  = lane & 15;
  const int koffA = (lane < 16) ? 0 : 8;    // 16-bit A layout half-wave K offset
  const int koffB = (lane < 16) ? 0 : 8;    // 16-bit B layout half-wave K offset

  v2u areg0, areg1, breg[4];

  // ---- prologue: load tile 0 and stage into lds[0] ----
  areg0 = *(gc_v2u*)(Ag + aoff0);
  areg1 = *(gc_v2u*)(Ag + aoff1);
#pragma unroll
  for (int j = 0; j < 4; ++j) breg[j] = *(gc_v2u*)(Bg + boff[j]);

  *(v2u*)(lds_a[0] + aq0 * 4) = areg0;
  *(v2u*)(lds_a[0] + aq1 * 4) = areg1;
#pragma unroll
  for (int j = 0; j < 4; ++j) {
    unsigned short* p = lds_b[0] + bkk[j];
    p[(bnn[j] + 0) * BK] = (unsigned short)(breg[j][0] & 0xFFFFu);
    p[(bnn[j] + 1) * BK] = (unsigned short)(breg[j][0] >> 16);
    p[(bnn[j] + 2) * BK] = (unsigned short)(breg[j][1] & 0xFFFFu);
    p[(bnn[j] + 3) * BK] = (unsigned short)(breg[j][1] >> 16);
  }
  __syncthreads();

  const unsigned int bstep = (unsigned)BK * ldb;
  int cur = 0;
  for (int k0 = 0; k0 < K; k0 += BK) {
    const bool has_next = (k0 + BK) < K;
    if (has_next) {
      aoff0 += BK; aoff1 += BK;
      areg0 = *(gc_v2u*)(Ag + aoff0);
      areg1 = *(gc_v2u*)(Ag + aoff1);
#pragma unroll
      for (int j = 0; j < 4; ++j) {
        boff[j] += bstep;
        breg[j] = *(gc_v2u*)(Bg + boff[j]);
      }
    }

    // ---- fragments from lds[cur] (packed dwords per ISA 16-bit layouts) ----
    FragAB afr[2], bfr[2];
#pragma unroll
    for (int s = 0; s < 2; ++s) {
      const unsigned short* ap = lds_a[cur] + (wm + s * 16 + mrow) * BK;
#pragma unroll
      for (int r = 0; r < 8; ++r) {
        int kk = (r < 4) ? (koffA + 2 * r) : (16 + koffA + 2 * (r - 4));
        afr[s].u[r] = *(const unsigned int*)(ap + kk);
      }
      const unsigned short* bp = lds_b[cur] + (wn + s * 16 + (lane & 15)) * BK;
#pragma unroll
      for (int r = 0; r < 8; ++r) {
        bfr[s].u[r] = *(const unsigned int*)(bp + ((r < 4) ? (koffB + 2 * r) : (16 + koffB + 2 * (r - 4))));
      }
    }

#pragma unroll
    for (int i = 0; i < 2; ++i)
#pragma unroll
      for (int j = 0; j < 2; ++j)
        acc[i][j] = __builtin_amdgcn_wmma_f32_16x16x32_bf16(
            false, afr[i].v, false, bfr[j].v, (short)0, acc[i][j], false, false);

    if (has_next) {
      int nxt = cur ^ 1;
      *(v2u*)(lds_a[nxt] + aq0 * 4) = areg0;
      *(v2u*)(lds_a[nxt] + aq1 * 4) = areg1;
#pragma unroll
      for (int j = 0; j < 4; ++j) {
        unsigned short* p = lds_b[nxt] + bkk[j];
        p[(bnn[j] + 0) * BK] = (unsigned short)(breg[j][0] & 0xFFFFu);
        p[(bnn[j] + 1) * BK] = (unsigned short)(breg[j][0] >> 16);
        p[(bnn[j] + 2) * BK] = (unsigned short)(breg[j][1] & 0xFFFFu);
        p[(bnn[j] + 3) * BK] = (unsigned short)(breg[j][1] >> 16);
      }
    }
    __syncthreads();
    cur ^= 1;
  }

  // ---- epilogue (no bounds checks: dims exact by construction) ----
  gc_float* biasg = (gc_float*)bias;
  g_float*  outf  = (g_float*)outb;
  g_ushort* outh  = (g_ushort*)outb;
  const int col0   = lane & 15;
  const int rowadd = (lane < 16) ? 0 : 8;
#pragma unroll
  for (int i = 0; i < 2; ++i) {
#pragma unroll
    for (int j = 0; j < 2; ++j) {
      FragD d; d.v = acc[i][j];
      unsigned int gn = bn0 + wn + j * 16 + col0;
#pragma unroll
      for (int r = 0; r < 8; ++r) {
        unsigned int gm = bm0 + wm + i * 16 + rowadd + r;
        float v = d.f[r] * scale;
        if (bias) v += biasg[gm];
        if (mode == 0) {
          outf[gm * (unsigned)ldc + gn] = v;
        } else {
          outh[gn * (unsigned)ldc + gm] = f32_to_bf16(v);
        }
      }
    }
  }
}

// ---------------- f32 -> bf16 convert (flat, for weights) ----------------
__global__ __launch_bounds__(256) void cvt_f32_bf16(
    const float* __restrict__ src, unsigned short* __restrict__ dst, long long n)
{
  long long i = (long long)blockIdx.x * 256 + threadIdx.x;
  long long stride = (long long)gridDim.x * 256;
  for (; i < n; i += stride) dst[i] = f32_to_bf16(src[i]);
}

// ---------------- x: f32 [.,T] -> bf16 [.,Tp] with zero pad ----------------
__global__ __launch_bounds__(256) void pad_cvt_x(
    const float* __restrict__ src, unsigned short* __restrict__ dst, long long rows)
{
  long long total = rows * kTp;
  long long i = (long long)blockIdx.x * 256 + threadIdx.x;
  long long stride = (long long)gridDim.x * 256;
  for (; i < total; i += stride) {
    long long row = i / kTp;
    int t = (int)(i % kTp);
    dst[i] = (t < kT) ? f32_to_bf16(src[row * kT + t]) : (unsigned short)0;
  }
}

// ---------------- 2x2 maxpool (+optional transpose) f32 -> bf16, zero-pad P --
__global__ __launch_bounds__(256) void pool_cvt(
    const float* __restrict__ src, unsigned short* __restrict__ dst,
    int transpose_out, long long strideIn, long long strideOut)
{
  const float* s = src + (long long)blockIdx.z * strideIn;
  unsigned short* d = dst + (long long)blockIdx.z * strideOut;
  int idx = blockIdx.x * 256 + threadIdx.x;
  if (idx >= kCI * kPp) return;
  int ci = idx / kPp;
  int p  = idx % kPp;
  unsigned short v = 0;
  if (p < kP) {
    int ph = p / kWP, pw = p % kWP;
    const float* row = s + (long long)ci * kTp + (2 * ph) * kW + 2 * pw;
    v = f32_to_bf16(fmaxf(fmaxf(row[0], row[1]), fmaxf(row[kW], row[kW + 1])));
  }
  if (transpose_out) d[(long long)p * kCI + ci] = v;   // [Pp x CI]
  else               d[(long long)ci * kPp + p] = v;   // [CI x Pp]
}

// ---------------- softmax over rows (len P, stored with stride Pp) ----------
__global__ __launch_bounds__(256) void softmax_rows(
    const float* __restrict__ f, unsigned short* __restrict__ a)
{
  long long row = blockIdx.x;
  const float* src = f + row * (long long)kPp;
  unsigned short* dst = a + row * (long long)kPp;
  __shared__ float red[256];
  int tid = threadIdx.x;

  float lmax = -3.4e38f;
  for (int i = tid; i < kP; i += 256) lmax = fmaxf(lmax, src[i]);
  red[tid] = lmax; __syncthreads();
  for (int s = 128; s > 0; s >>= 1) {
    if (tid < s) red[tid] = fmaxf(red[tid], red[tid + s]);
    __syncthreads();
  }
  float rmax = red[0]; __syncthreads();

  float lsum = 0.0f;
  for (int i = tid; i < kP; i += 256) lsum += __expf(src[i] - rmax);
  red[tid] = lsum; __syncthreads();
  for (int s = 128; s > 0; s >>= 1) {
    if (tid < s) red[tid] += red[tid + s];
    __syncthreads();
  }
  float rinv = 1.0f / red[0];
  for (int i = tid; i < kP; i += 256)
    dst[i] = f32_to_bf16(__expf(src[i] - rmax) * rinv);
  // zero the K-padding tail (a is the contraction operand of the next GEMM)
  for (int i = kP + tid; i < kPp; i += 256) dst[i] = 0;
}

// ---------------- BN statistics: one block per channel ----------------
__global__ __launch_bounds__(256) void bn_stats(
    const float* __restrict__ z, float* __restrict__ mean, float* __restrict__ var)
{
  int c = blockIdx.x;
  __shared__ float s1[256], s2[256];
  int tid = threadIdx.x;
  float a = 0.0f, b = 0.0f;
  const long long tot = (long long)kN * kT;
  for (long long i = tid; i < tot; i += 256) {
    long long nn = i / kT;
    int t = (int)(i % kT);
    float v = z[(nn * kC + c) * (long long)kTp + t];
    a += v; b += v * v;
  }
  s1[tid] = a; s2[tid] = b; __syncthreads();
  for (int s = 128; s > 0; s >>= 1) {
    if (tid < s) { s1[tid] += s1[tid + s]; s2[tid] += s2[tid + s]; }
    __syncthreads();
  }
  if (tid == 0) {
    float inv = 1.0f / (float)tot;
    float m = s1[0] * inv;
    mean[c] = m;
    var[c]  = s2[0] * inv - m * m;
  }
}

// ---------------- BN normalize + affine + residual ----------------
__global__ __launch_bounds__(256) void bn_residual(
    const float* __restrict__ z, const float* __restrict__ x,
    const float* __restrict__ mean, const float* __restrict__ var,
    const float* __restrict__ gamma, const float* __restrict__ beta,
    float* __restrict__ out, long long total)
{
  long long i = (long long)blockIdx.x * 256 + threadIdx.x;
  long long stride = (long long)gridDim.x * 256;
  for (; i < total; i += stride) {
    long long nc = i / kT;
    int t = (int)(i % kT);
    int c = (int)(nc % kC);
    float zv = z[nc * kTp + t];
    float zn = (zv - mean[c]) * rsqrtf(var[c] + 1e-5f);
    out[i] = zn * gamma[c] + beta[c] + x[i];
  }
}

// ---------------- host orchestration ----------------
extern "C" void kernel_launch(void* const* d_in, const int* in_sizes, int n_in,
                              void* d_out, int out_size, void* d_ws, size_t ws_size,
                              hipStream_t stream) {
  const float* x       = (const float*)d_in[0];
  const float* w_theta = (const float*)d_in[1];
  const float* b_theta = (const float*)d_in[2];
  const float* w_phi   = (const float*)d_in[3];
  const float* b_phi   = (const float*)d_in[4];
  const float* w_g     = (const float*)d_in[5];
  const float* b_g     = (const float*)d_in[6];
  const float* w_z     = (const float*)d_in[7];
  const float* b_z     = (const float*)d_in[8];
  const float* gamma   = (const float*)d_in[9];
  const float* beta    = (const float*)d_in[10];
  float* out = (float*)d_out;

  char* ws = (char*)d_ws;
  size_t off = 0;
  auto carve = [&](size_t bytes) -> void* {
    void* p = ws + off;
    off = (off + bytes + 255) & ~(size_t)255;
    return p;
  };

  unsigned short* x_bf    = (unsigned short*)carve((size_t)kN * kC * kTp * 2);  // 26.2 MB
  unsigned short* wth_bf  = (unsigned short*)carve((size_t)kCI * kC * 2);
  unsigned short* wph_bf  = (unsigned short*)carve((size_t)kCI * kC * 2);
  unsigned short* wg_bf   = (unsigned short*)carve((size_t)kCI * kC * 2);
  unsigned short* wz_bf   = (unsigned short*)carve((size_t)kC * kCI * 2);
  unsigned short* theta_t = (unsigned short*)carve((size_t)kN * kTp * kCI * 2); // 13.1 MB
  unsigned short* phi_pl  = (unsigned short*)carve((size_t)kN * kCI * kPp * 2); // 3.7 MB
  unsigned short* g_plT   = (unsigned short*)carve((size_t)kN * kPp * kCI * 2); // 3.7 MB
  // region1: phi_full + g_full (f32 [CI x Tp]), later reused as a_bf
  char* region1 = (char*)carve(2 * (size_t)kN * kCI * kTp * 4);                 // 52.4 MB
  float* phi_full = (float*)region1;
  float* g_full   = (float*)(region1 + (size_t)kN * kCI * kTp * 4);
  unsigned short* a_bf = (unsigned short*)region1;  // N*T*Pp*2 = 45 MB, fits
  // region2: f (f32 [T x Pp]), later reused as z (f32 [C x Tp])
  char* region2 = (char*)carve((size_t)kN * kT * kPp * 4);                      // 89.9 MB
  float* f_log = (float*)region2;
  float* z_buf = (float*)region2;                   // N*C*Tp*4 = 52.4 MB, fits
  unsigned short* y_bf = (unsigned short*)carve((size_t)kN * kCI * kTp * 2);    // 13.1 MB
  float* mean_c = (float*)carve(kC * 4);
  float* var_c  = (float*)carve(kC * 4);
  (void)ws_size; (void)n_in; (void)in_sizes; (void)out_size;

  // 1) convert inputs to bf16 (x gets zero-padded columns to Tp)
  pad_cvt_x<<<dim3(4096), dim3(256), 0, stream>>>(x, x_bf, (long long)kN * kC);
  cvt_f32_bf16<<<dim3(512), dim3(256), 0, stream>>>(w_theta, wth_bf, (long long)kCI * kC);
  cvt_f32_bf16<<<dim3(512), dim3(256), 0, stream>>>(w_phi,   wph_bf, (long long)kCI * kC);
  cvt_f32_bf16<<<dim3(512), dim3(256), 0, stream>>>(w_g,     wg_bf,  (long long)kCI * kC);
  cvt_f32_bf16<<<dim3(512), dim3(256), 0, stream>>>(w_z,     wz_bf,  (long long)kC * kCI);

  // 2) input convs: [CI,C] x [C,Tp] per batch
  dim3 gconv(kTp / BN, kCI / BM, kN);   // 25 x 4 x 8
  gemm_bf16_wmma<<<gconv, dim3(256), 0, stream>>>(
      wth_bf, x_bf, b_theta, theta_t,
      kCI, kTp, kC, kC, kTp, kCI, 1.0f, /*mode*/2,
      0LL, (long long)kC * kTp, (long long)kTp * kCI * 2);
  gemm_bf16_wmma<<<gconv, dim3(256), 0, stream>>>(
      wph_bf, x_bf, b_phi, phi_full,
      kCI, kTp, kC, kC, kTp, kTp, 1.0f, /*mode*/0,
      0LL, (long long)kC * kTp, (long long)kCI * kTp * 4);
  gemm_bf16_wmma<<<gconv, dim3(256), 0, stream>>>(
      wg_bf, x_bf, b_g, g_full,
      kCI, kTp, kC, kC, kTp, kTp, 1.0f, /*mode*/0,
      0LL, (long long)kC * kTp, (long long)kCI * kTp * 4);

  // 3) 2x2 maxpool + bf16, zero-padded to Pp (g stored transposed [Pp x CI])
  dim3 gpool((kCI * kPp + 255) / 256, 1, kN);
  pool_cvt<<<gpool, dim3(256), 0, stream>>>(
      phi_full, phi_pl, 0, (long long)kCI * kTp, (long long)kCI * kPp);
  pool_cvt<<<gpool, dim3(256), 0, stream>>>(
      g_full, g_plT, 1, (long long)kCI * kTp, (long long)kPp * kCI);

  // 4) f = theta^T * phi * ci^-0.5   [T x Pp] f32
  dim3 gf(kPp / BN, kT / BM, kN);       // 7 x 49 x 8
  gemm_bf16_wmma<<<gf, dim3(256), 0, stream>>>(
      theta_t, phi_pl, nullptr, f_log,
      kT, kPp, kCI, kCI, kPp, kPp, 0.0625f /*1/sqrt(256)*/, /*mode*/0,
      (long long)kTp * kCI, (long long)kCI * kPp, (long long)kT * kPp * 4);

  // 5) softmax over P, bf16 attention with zeroed K-pad tail
  softmax_rows<<<dim3(kN * kT), dim3(256), 0, stream>>>(f_log, a_bf);

  // 6) y^T = a * g^T -> transposed bf16 store y[ci][t], K = Pp (zero padded)
  dim3 gy(kCI / BN, kT / BM, kN);       // 2 x 49 x 8
  gemm_bf16_wmma<<<gy, dim3(256), 0, stream>>>(
      a_bf, g_plT, nullptr, y_bf,
      kT, kCI, kPp, kPp, kCI, kTp, 1.0f, /*mode*/2,
      (long long)kT * kPp, (long long)kPp * kCI, (long long)kCI * kTp * 2);

  // 7) z = w_z * y + b_z   [C x Tp] f32
  dim3 gz(kTp / BN, kC / BM, kN);       // 25 x 8 x 8
  gemm_bf16_wmma<<<gz, dim3(256), 0, stream>>>(
      wz_bf, y_bf, b_z, z_buf,
      kC, kTp, kCI, kCI, kTp, kTp, 1.0f, /*mode*/0,
      0LL, (long long)kCI * kTp, (long long)kC * kTp * 4);

  // 8) batch-norm statistics + fused normalize/affine/residual
  bn_stats<<<dim3(kC), dim3(256), 0, stream>>>(z_buf, mean_c, var_c);
  bn_residual<<<dim3(4096), dim3(256), 0, stream>>>(
      z_buf, x, mean_c, var_c, gamma, beta, out, (long long)kN * kC * kT);
}
